// TnoFD_84679575208687
// MI455X (gfx1250) — matrix operand order repaired and checked by reading, not compiled, not run on hardware
//
#include <hip/hip_runtime.h>

// ---------------- problem dimensions (fixed by the reference) ----------------
#define BB    16
#define HH    8
#define NN    4096
#define DD    64
#define NF    4097            // rfft bins of fft_size=8192
#define CH    (HH*DD)         // 512 channels
#define RPE   64
#define ODIM  512             // H*D
#define PI_F  3.14159265358979f

// ---------------- CDNA5 WMMA vector types (probe-verified signatures) -------
typedef __attribute__((ext_vector_type(16))) __bf16 v16bf;
typedef __attribute__((ext_vector_type(8)))  __bf16 v8bf;
typedef __attribute__((ext_vector_type(8)))  float  v8f;

// ============================================================================
// Stage 1: RPE MLP  res[f][h*D+d], f in [0,4097)
//   hdn = w_f*pos_w + pos_b ; 3x (relu(rmsnorm(h)) @ W + b) ; out layer.
// Tiny (0.4 GFLOP): fp32 VALU, weights staged in LDS, 4 rows per 256-thr block.
// ============================================================================
__global__ __launch_bounds__(256) void rpe_mlp_kernel(
    const float* __restrict__ pos_w, const float* __restrict__ pos_b,
    const float* __restrict__ w1, const float* __restrict__ b1,
    const float* __restrict__ w2, const float* __restrict__ b2,
    const float* __restrict__ w3, const float* __restrict__ b3,
    const float* __restrict__ out_w, const float* __restrict__ out_b,
    float* __restrict__ res)
{
    __shared__ float sW[3][RPE*RPE];
    __shared__ float sB[3][RPE];
    __shared__ float sPw[RPE], sPb[RPE], sOb[ODIM];
    __shared__ float sg[4][RPE];     // post relu/norm activations per group
    __shared__ float sr[4][RPE];     // reduction scratch per group

    const int tid = threadIdx.x;
    for (int i = tid; i < RPE*RPE; i += 256) {
        sW[0][i] = w1[i]; sW[1][i] = w2[i]; sW[2][i] = w3[i];
    }
    if (tid < RPE) {
        sB[0][tid] = b1[tid]; sB[1][tid] = b2[tid]; sB[2][tid] = b3[tid];
        sPw[tid] = pos_w[tid]; sPb[tid] = pos_b[tid];
    }
    for (int i = tid; i < ODIM; i += 256) sOb[i] = out_b[i];
    __syncthreads();

    const int grp = tid >> 6;        // 4 groups of 64 threads, one f each
    const int j   = tid & 63;

    for (int it = 0; it < 17; ++it) {               // 17*256 = 4352 >= 4097
        const int  f     = it*256 + (int)blockIdx.x*4 + grp;
        const bool valid = (f < NF);
        const float wf   = PI_F * (float)f / 4096.0f;   // linspace(0,pi,4097)
        float h = wf * sPw[j] + sPb[j];

        for (int L = 0; L < 4; ++L) {               // 3 hidden + final norm
            sr[grp][j] = h*h;
            __syncthreads();
            for (int s = 32; s > 0; s >>= 1) {
                if (j < s) sr[grp][j] += sr[grp][j+s];
                __syncthreads();
            }
            float rms = sqrtf(sr[grp][0] * (1.0f/RPE));
            float gv  = h / (rms + 1e-8f);
            gv = gv > 0.0f ? gv : 0.0f;             // relu(rmsnorm(h))
            sg[grp][j] = gv;
            __syncthreads();
            if (L < 3) {
                float acc = sB[L][j];
                #pragma unroll 8
                for (int k = 0; k < RPE; ++k) acc += sg[grp][k] * sW[L][k*RPE + j];
                h = acc;
                __syncthreads();
            }
        }
        // output layer: 512 outputs, 8 per thread (out_w streamed from L2)
        for (int o = j; o < ODIM; o += 64) {
            float acc = sOb[o];
            #pragma unroll 8
            for (int k = 0; k < RPE; ++k) acc += sg[grp][k] * out_w[k*ODIM + o];
            if (valid) res[(size_t)f*ODIM + o] = acc;
        }
        __syncthreads();
    }
}

// ============================================================================
// Stage 2: causal kernel taps via real cosine transform, as bf16 WMMA matmul:
//   k[m][n] = c_m * (2/8192) * sum_{j=0}^{4096} w_j * res[j][n] * cos(pi*m*j/4096)
//   w_0 = w_4096 = 1, else 2;  c_0 = 0.5, else 1.  (Nyquist (-1)^m = cos.)
// A-tiles (cos) generated on the fly; res prescaled while staging to LDS.
// Grid (32,4): 128 m-rows x 128 n-cols per 8-wave workgroup.
// ============================================================================
__global__ __launch_bounds__(256) void rpe_dft_kernel(
    const float* __restrict__ res,        // [4097][512] fp32
    __bf16* __restrict__ kf)              // [512][4096] bf16 taps out
{
    __shared__ __align__(16) __bf16 rT[128*40];   // res^T chunk: [n_loc][jj], pad->40

    const int tid  = threadIdx.x;
    const int wave = tid >> 5, lane = tid & 31, ln = lane & 15;
    const bool hi  = lane >= 16;
    const int khalf  = hi ? 8 : 0;     // unused; A and B both take K = koff + e
    const int koff   = hi ? 16 : 0;    // B element->K offset
    const int rowoff = hi ? 8 : 0;     // D row offset
    const int m0 = (int)blockIdx.x*128 + wave*16;
    const int n0 = (int)blockIdx.y*128;
    const float invF = 2.0f/8192.0f;

    v8f acc[8];
    #pragma unroll
    for (int t = 0; t < 8; ++t) acc[t] = (v8f){0,0,0,0,0,0,0,0};

    for (int jc = 0; jc < 129; ++jc) {            // 129*32 = 4128 >= 4097
        // stage scaled res chunk (32 j x 128 n), transposed for b128 B-loads
        #pragma unroll
        for (int r = 0; r < 16; ++r) {
            int e2 = r*256 + tid;
            int jj = e2 >> 7, nl = e2 & 127;
            int jg = jc*32 + jj;
            float v = 0.0f;
            if (jg < NF) {
                v = res[(size_t)jg*ODIM + n0 + nl];
                v *= (jg == 0 || jg == NF-1) ? invF : 2.0f*invF;
            }
            rT[nl*40 + jj] = (__bf16)v;
        }
        __syncthreads();

        // A tile: cos(pi*m*j/4096), integer-reduced angle so __cosf stays fast
        const int m = m0 + ln;
        v16bf a;
        #pragma unroll
        for (int e = 0; e < 16; ++e) {
            int K  = koff + e;
            int jg = jc*32 + K;
            int rr = (m * jg) & 8191;             // period 8192 of cos(pi*q/4096)
            a[e] = (__bf16)__cosf((float)rr * (PI_F/4096.0f));
        }
        #pragma unroll
        for (int nt = 0; nt < 8; ++nt) {
            const __bf16* bs = &rT[(nt*16 + ln)*40 + koff];   // 16B aligned
            v8bf b0 = *(const v8bf*)bs;
            v8bf b1 = *(const v8bf*)(bs + 8);
            v16bf bb;
            #pragma unroll
            for (int e = 0; e < 8; ++e) { bb[e] = b0[e]; bb[e+8] = b1[e]; }
            acc[nt] = __builtin_amdgcn_wmma_f32_16x16x32_bf16(
                false, a, false, bb, (short)0, acc[nt], false, false);
        }
        __syncthreads();
    }

    // write D tiles -> taps kf[channel][m] (bf16), halve the m==0 tap
    #pragma unroll
    for (int nt = 0; nt < 8; ++nt) {
        const int n = n0 + nt*16 + ln;
        #pragma unroll
        for (int v = 0; v < 8; ++v) {
            int m = m0 + v + rowoff;
            float val = acc[nt][v];
            if (m == 0) val *= 0.5f;
            kf[(size_t)n*NN + m] = (__bf16)val;
        }
    }
}

// ============================================================================
// Stage 3: per-channel causal convolution as lower-triangular Toeplitz matmul
//   OUT(4096x16) = K(tri-Toeplitz from taps) @ X(4096x16)  (N dim == batch B)
// One workgroup (8 waves) per channel; X^T (16x4096 bf16) + taps (+32-zero
// guard) live in LDS (~140KB of 320KB) so x is read from HBM exactly once.
//
// Diagonal-shared A tiles: A(it,jt) depends only on delta = 16*it - 32*jt, so
// the 8 same-parity row tiles {it0+2k} at jt = s+k share one A. Each wave
// carries 8 f32 accumulators; per A build (16 ds_load_u16, amortized /8) it
// issues 8 independent v_wmma_f32_16x16x32_bf16 (good XDL pipelining, no
// D->A/B hazards) each fed by 2 ds_load_b128 -> ~4 DS instr per WMMA.
// Wave w takes diagonal groups g = w and 15-w (both parities): exactly
// uniform WMMA count per wave.
// ============================================================================
#define XT_STRIDE 4104          // 4096 + 8 halfs pad: keeps rows 16B aligned,
                                // staggers banks; 16*4104*2 = 131328 B
__global__ __launch_bounds__(256) void toeplitz_conv_kernel(
    const float* __restrict__ x,          // [B][H][N][D] fp32
    const __bf16* __restrict__ kf,        // [512][4096] bf16 taps
    float* __restrict__ out)              // [B][H][N][D] fp32
{
    extern __shared__ __align__(16) __bf16 smem[];
    __bf16* Xt = smem;                    // [16][XT_STRIDE]  X^T in bf16
    __bf16* kl = smem + 16*XT_STRIDE;     // [32 zero guard + 4096 taps]

    const int c = blockIdx.x;             // channel = h*64 + d
    const int h = c >> 6, d = c & 63;
    const int tid = threadIdx.x;

    {   // stage X^T: Xt[b][j] = x[b][h][j][d]  (cross-WG L2 reuse over d)
        const int b  = tid & 15;
        const int j0 = tid >> 4;
        const size_t xb = (size_t)b*(HH*NN*DD) + (size_t)h*(NN*DD) + d;
        for (int j = j0; j < NN; j += 16)
            Xt[b*XT_STRIDE + j] = (__bf16)x[xb + (size_t)j*DD];
    }
    if (tid < 32) kl[tid] = (__bf16)0.0f;                   // negative-tap guard
    for (int m = tid; m < NN; m += 256) kl[32 + m] = kf[(size_t)c*NN + m];
    __syncthreads();

    const int wave = tid >> 5, lane = tid & 31, ln = lane & 15;
    const bool hi  = lane >= 16;
    const int khalf  = hi ? 8 : 0;
    const int koff   = hi ? 16 : 0;
    const int rowoff = hi ? 8 : 0;
    const size_t obase = (size_t)ln*(HH*NN*DD) + (size_t)h*(NN*DD) + d; // b=ln

    for (int t = 0; t < 4; ++t) {
        const int g    = (t & 1) ? (15 - wave) : wave;  // diagonal group 0..15
        const int p    = t >> 1;                        // row-tile parity
        const int it0  = 16*g + p;                      // first of 8 row tiles
        const int smax = it0 >> 1;
        const int ibase = it0*16 + ln;

        v8f acc[8];
        #pragma unroll
        for (int k = 0; k < 8; ++k) acc[k] = (v8f){0,0,0,0,0,0,0,0};

        // A[r][c] = tap[16*it0 + r - 32*s - c]; lane ln holds row M=ln,
        // element e -> K = koff+e (contiguous k order, matches loadB).
        auto makeA = [&](int s, v16bf& a) {
            const int t0 = ibase - 32*s - koff;
            #pragma unroll
            for (int e = 0; e < 8; ++e)  a[e] = kl[32 + t0 - e];
            #pragma unroll
            for (int e = 8; e < 16; ++e) a[e] = kl[32 + t0 - e];
        };
        // B[k][n] = X[jt*32+k][n]: 16 consecutive halfs per lane -> 2x b128
        auto loadB = [&](int jt, v16bf& bb) {
            const __bf16* bs = &Xt[ln*XT_STRIDE + jt*32 + koff];
            v8bf b0 = *(const v8bf*)bs;
            v8bf b1 = *(const v8bf*)(bs + 8);
            #pragma unroll
            for (int e = 0; e < 8; ++e) { bb[e] = b0[e]; bb[e+8] = b1[e]; }
        };

        // prologue: s in [-7,-1]; tile k active iff jt = s+k >= 0.
        // (s,k compile-time constants -> guards fold away)
        #pragma unroll
        for (int s = -7; s <= -1; ++s) {
            v16bf a; makeA(s, a);
            #pragma unroll
            for (int k = 0; k < 8; ++k) {
                if (k + s >= 0) {
                    v16bf bb; loadB(s + k, bb);
                    acc[k] = __builtin_amdgcn_wmma_f32_16x16x32_bf16(
                        false, a, false, bb, (short)0, acc[k], false, false);
                }
            }
        }
        // main diagonal sweep: one shared A feeds 8 independent WMMAs
        for (int s = 0; s <= smax; ++s) {
            v16bf a; makeA(s, a);
            #pragma unroll
            for (int k = 0; k < 8; ++k) {
                v16bf bb; loadB(s + k, bb);
                acc[k] = __builtin_amdgcn_wmma_f32_16x16x32_bf16(
                    false, a, false, bb, (short)0, acc[k], false, false);
            }
        }
        // D tiles: VGPR v -> row (it0+2k)*16 + v + rowoff, col (batch) = ln
        #pragma unroll
        for (int k = 0; k < 8; ++k) {
            const int mrow = (it0 + 2*k)*16 + rowoff;
            #pragma unroll
            for (int v = 0; v < 8; ++v)
                out[obase + (size_t)(mrow + v)*DD] = acc[k][v];
        }
    }
}

// ============================================================================
// Host entry
// ============================================================================
extern "C" void kernel_launch(void* const* d_in, const int* in_sizes, int n_in,
                              void* d_out, int out_size, void* d_ws, size_t ws_size,
                              hipStream_t stream) {
    (void)in_sizes; (void)n_in; (void)out_size; (void)ws_size;
    const float* x    = (const float*)d_in[0];
    const float* posw = (const float*)d_in[1];
    const float* posb = (const float*)d_in[2];
    const float* w1   = (const float*)d_in[3];
    const float* b1   = (const float*)d_in[4];
    const float* w2   = (const float*)d_in[5];
    const float* b2   = (const float*)d_in[6];
    const float* w3   = (const float*)d_in[7];
    const float* b3   = (const float*)d_in[8];
    const float* outw = (const float*)d_in[9];
    const float* outb = (const float*)d_in[10];

    // scratch layout: res (4097*512 fp32 = 8.39MB) | taps (512*4096 bf16 = 4MB)
    float*  res = (float*)d_ws;
    __bf16* kf  = (__bf16*)((char*)d_ws + (size_t)NF*ODIM*sizeof(float));

    rpe_mlp_kernel<<<64, 256, 0, stream>>>(posw, posb, w1, b1, w2, b2, w3, b3,
                                           outw, outb, res);

    dim3 g2(32, 4);
    rpe_dft_kernel<<<g2, 256, 0, stream>>>(res, kf);

    const size_t lds = (size_t)(16*XT_STRIDE + 32 + NN) * sizeof(__bf16); // ~140KB
    toeplitz_conv_kernel<<<CH, 256, lds, stream>>>(x, kf, (float*)d_out);
}
